// MLPEncoder_1889785610578
// MI455X (gfx1250) — matrix operand order repaired and checked by reading, not compiled, not run on hardware
//
#include <hip/hip_runtime.h>
#include <hip/hip_bf16.h>

// Problem constants (match reference)
#define N_NODES 20000
#define N_EDGES 400000
#define FN 16
#define FE 16
#define HID 128
#define D 16
#define TWO_D 32
#define ET 32                      // edges per wave-tile (2 x 16-edge sub-tiles)
#define NT_EDGE (N_EDGES / ET)     // 12500

typedef __attribute__((ext_vector_type(16))) _Float16 v16h;
typedef __attribute__((ext_vector_type(8)))  _Float16 v8h;
typedef __attribute__((ext_vector_type(8)))  float    v8f;

// f16 weight blob layout (halves): W1 padded [32][128] | b2 [32][16] | W2 [128][512]
#define BLOB_W1_OFF 0
#define BLOB_B2_OFF (32 * 128)
#define BLOB_W2_OFF (32 * 128 + 32 * 16)
#define BLOB_HALVES (32 * 128 + 32 * 16 + 128 * 512)   // 70144
#define BLOB_BYTES  (BLOB_HALVES * 2)                  // 140288 (16B multiple)

// per-wave LDS: u[2][16][128]f16 (8192) + ea[2][16][32]f16 (2048)
//             + nf16[2][16][32]f16 (2048) + nf32[2][16][32]f32 (4096) + dst[32] (128)
#define PER_WAVE_B 16512
#define SMEM_BYTES (BLOB_BYTES + 4 * PER_WAVE_B)       // 206336 < 320KB/WGP

// ---------------------------------------------------------------------------
// Utility kernels (tiny, bandwidth-trivial)
// ---------------------------------------------------------------------------
__global__ void zero_kernel(float* __restrict__ p, int n) {
    int i = blockIdx.x * blockDim.x + threadIdx.x;
    if (i < n) p[i] = 0.f;
}

__global__ void idx_copy_kernel(const int* __restrict__ eidx, float* __restrict__ out) {
    int i = blockIdx.x * blockDim.x + threadIdx.x;
    if (i < 2 * N_EDGES) out[i] = (float)eidx[i];
}

// Pre-convert shared-MLP weights to one contiguous f16 blob in workspace.
__global__ void weights_prep_kernel(const float* __restrict__ W1,
                                    const float* __restrict__ b2,
                                    const float* __restrict__ W2,
                                    _Float16* __restrict__ blob) {
    int i = blockIdx.x * blockDim.x + threadIdx.x;
    if (i >= BLOB_HALVES) return;
    float v;
    if (i < BLOB_B2_OFF) {                       // W1, K-padded 16->32 with zeros
        v = (i < FE * HID) ? W1[i] : 0.f;
    } else if (i < BLOB_W2_OFF) {                // b2 as [32][16]
        v = b2[i - BLOB_B2_OFF];
    } else {                                     // W2 [128][512]
        v = W2[i - BLOB_W2_OFF];
    }
    blob[i] = (_Float16)v;
}

// h0 = relu(x @ Wn + bn)   [N,16]
__global__ void node_embed_kernel(const float* __restrict__ x,
                                  const float* __restrict__ Wn,
                                  const float* __restrict__ bn,
                                  float* __restrict__ h) {
    int n = blockIdx.x * blockDim.x + threadIdx.x;
    if (n >= N_NODES) return;
    float xr[FN];
#pragma unroll
    for (int k = 0; k < FN; ++k) xr[k] = x[(size_t)n * FN + k];
#pragma unroll
    for (int d = 0; d < D; ++d) {
        float s = bn[d];
#pragma unroll
        for (int k = 0; k < FN; ++k) s += xr[k] * Wn[k * D + d];
        h[(size_t)n * D + d] = fmaxf(s, 0.f);
    }
}

// hout = aggr + hin @ root + bias   [N,16]
__global__ void node_update_kernel(const float* __restrict__ hin,
                                   const float* __restrict__ aggr,
                                   const float* __restrict__ root,
                                   const float* __restrict__ bias,
                                   float* __restrict__ hout) {
    int n = blockIdx.x * blockDim.x + threadIdx.x;
    if (n >= N_NODES) return;
    float hr[D];
#pragma unroll
    for (int k = 0; k < D; ++k) hr[k] = hin[(size_t)n * D + k];
#pragma unroll
    for (int d = 0; d < D; ++d) {
        float s = aggr[(size_t)n * D + d] + bias[d];
#pragma unroll
        for (int k = 0; k < D; ++k) s += hr[k] * root[k * D + d];
        hout[(size_t)n * D + d] = s;
    }
}

// ee = relu(ea @ We + be); out_ls = log_softmax(ee)
__global__ void edge_embed_kernel(const float* __restrict__ ea,
                                  const float* __restrict__ We,
                                  const float* __restrict__ be,
                                  float* __restrict__ out_ee,
                                  float* __restrict__ out_ls) {
    int e = blockIdx.x * blockDim.x + threadIdx.x;
    if (e >= N_EDGES) return;
    float ar[FE];
#pragma unroll
    for (int k = 0; k < FE; ++k) ar[k] = ea[(size_t)e * FE + k];
    float v[D];
#pragma unroll
    for (int d = 0; d < D; ++d) {
        float s = be[d];
#pragma unroll
        for (int k = 0; k < FE; ++k) s += ar[k] * We[k * D + d];
        v[d] = fmaxf(s, 0.f);
    }
    float m = v[0];
#pragma unroll
    for (int d = 1; d < D; ++d) m = fmaxf(m, v[d]);
    float sum = 0.f;
#pragma unroll
    for (int d = 0; d < D; ++d) sum += __expf(v[d] - m);
    float lse = __logf(sum);
#pragma unroll
    for (int d = 0; d < D; ++d) {
        out_ee[(size_t)e * D + d] = v[d];
        out_ls[(size_t)e * D + d] = v[d] - m - lse;
    }
}

// ---------------------------------------------------------------------------
// Fused edge-conv kernel. One wave32 = 32 edges (2 x 16-edge WMMA sub-tiles;
// every LDS-resident B operand feeds two v_wmma ops).
//   u   = relu(ea_tile @ W1 + b1)                      (2x8 v_wmma f16 K=32)
//   msg = nf @ reshape(b2,[32,16])                     (2x  v_wmma: bias fold)
//   for k in 0..31:  w_k = u @ W2[:,16k:16k+16]        (2x4 v_wmma, K=128)
//                    msg[e,:] += nf[e,k] * w_k[e,:]
//   atomicAdd(aggr[dst[e]], msg[e])
// Weights staged once per persistent block into LDS via async-to-LDS copies.
// ---------------------------------------------------------------------------
__global__ void __launch_bounds__(128) edge_conv_kernel(
    const float*    __restrict__ h,
    const int*      __restrict__ eidx,
    const float*    __restrict__ ea,
    const _Float16* __restrict__ blob,   // f16 weights in workspace
    const float*    __restrict__ b1,
    float*          __restrict__ aggr) {
    extern __shared__ char smem[];
    _Float16* w1s = (_Float16*)smem;            // [32][128]
    _Float16* b2s = w1s + BLOB_B2_OFF;          // [32][16]
    _Float16* w2s = w1s + BLOB_W2_OFF;          // [128][512]
    char* wvbase  = smem + BLOB_BYTES;

    const int wave = threadIdx.x >> 5;
    const int lane = threadIdx.x & 31;
    char* my = wvbase + wave * PER_WAVE_B;
    _Float16* u_s    = (_Float16*)my;             // [2][16][128] f16
    _Float16* ea_s   = (_Float16*)(my + 8192);    // [2][16][32]  f16 (K-padded)
    _Float16* nf16_s = (_Float16*)(my + 10240);   // [2][16][32]  f16
    float*    nf_s   = (float*)(my + 12288);      // [2][16][32]  f32
    int*      dst_s  = (int*)(my + 16384);        // [32]

    // --- stage f16 weight blob into LDS with async-to-LDS DMA (once/block) ---
    {
        const uint32_t lds0 = (uint32_t)(uintptr_t)smem;   // WG-relative LDS offset
        const uint64_t g0   = (uint64_t)(uintptr_t)blob;
        for (int i = threadIdx.x; i < BLOB_BYTES / 16; i += blockDim.x) {
            uint32_t la = lds0 + (uint32_t)i * 16u;
            uint64_t ga = g0 + (uint64_t)i * 16u;
            asm volatile("global_load_async_to_lds_b128 %0, %1, off"
                         :: "v"(la), "v"(ga) : "memory");
        }
        asm volatile("s_wait_asynccnt 0x0" ::: "memory");
    }
    __syncthreads();

    const int eL   = lane & 15;          // row within a 16-edge sub-tile (A-matrix M)
    const int hsel = (lane >> 4) << 3;   // 0 / 8 : A-layout K-half select & acc M-offset

    // ea K-padding (cols 16..31) is constant zero: write once per kernel
    {
        _Float16* er = ea_s + lane * 32 + 16;    // lane -> row `lane` of [2][16]
#pragma unroll
        for (int j = 0; j < 16; ++j) er[j] = (_Float16)0.f;
    }

    // hoist the b2 B-tile (constant): col N = lane%16, K in table order
    v16h Bb2;
    {
#pragma unroll
        for (int j = 0; j < 8; ++j) { Bb2[j] = b2s[(hsel + j) * 16 + eL]; Bb2[8 + j] = b2s[(16 + hsel + j) * 16 + eL]; }
    }

    const int nWaves = (gridDim.x * blockDim.x) >> 5;
    const int waveG  = (blockIdx.x * blockDim.x + threadIdx.x) >> 5;

    for (int tile = waveG; tile < NT_EDGE; tile += nWaves) {
        const int e = tile * ET + lane;          // each lane owns one edge
        // prefetch next tile's gather inputs while this tile computes
        const int en = e + nWaves * ET;
        if (en < N_EDGES) {
            __builtin_prefetch(eidx + en, 0, 0);
            __builtin_prefetch(eidx + N_EDGES + en, 0, 0);
            __builtin_prefetch(ea + (size_t)en * FE, 0, 0);
        }
        // gather both endpoints of this lane's edge
        const int di = eidx[N_EDGES + e];        // dst -> nf cols 0..15
        const int si = eidx[e];                  // src -> nf cols 16..31
        const float4* dv = (const float4*)(h + (size_t)di * D);
        float4 d0 = dv[0], d1 = dv[1], d2 = dv[2], d3 = dv[3];
        const float4* sv = (const float4*)(h + (size_t)si * D);
        float4 s0 = sv[0], s1 = sv[1], s2 = sv[2], s3 = sv[3];

        float* nfr = nf_s + lane * TWO_D;        // row `lane` of [2][16][32]
        ((float4*)nfr)[0] = d0; ((float4*)nfr)[1] = d1;
        ((float4*)nfr)[2] = d2; ((float4*)nfr)[3] = d3;
        ((float4*)nfr)[4] = s0; ((float4*)nfr)[5] = s1;
        ((float4*)nfr)[6] = s2; ((float4*)nfr)[7] = s3;

        _Float16* np = nf16_s + lane * TWO_D;    // f16 copy for the b2-fold WMMA
        np[0]  = (_Float16)d0.x; np[1]  = (_Float16)d0.y;
        np[2]  = (_Float16)d0.z; np[3]  = (_Float16)d0.w;
        np[4]  = (_Float16)d1.x; np[5]  = (_Float16)d1.y;
        np[6]  = (_Float16)d1.z; np[7]  = (_Float16)d1.w;
        np[8]  = (_Float16)d2.x; np[9]  = (_Float16)d2.y;
        np[10] = (_Float16)d2.z; np[11] = (_Float16)d2.w;
        np[12] = (_Float16)d3.x; np[13] = (_Float16)d3.y;
        np[14] = (_Float16)d3.z; np[15] = (_Float16)d3.w;
        np[16] = (_Float16)s0.x; np[17] = (_Float16)s0.y;
        np[18] = (_Float16)s0.z; np[19] = (_Float16)s0.w;
        np[20] = (_Float16)s1.x; np[21] = (_Float16)s1.y;
        np[22] = (_Float16)s1.z; np[23] = (_Float16)s1.w;
        np[24] = (_Float16)s2.x; np[25] = (_Float16)s2.y;
        np[26] = (_Float16)s2.z; np[27] = (_Float16)s2.w;
        np[28] = (_Float16)s3.x; np[29] = (_Float16)s3.y;
        np[30] = (_Float16)s3.z; np[31] = (_Float16)s3.w;

        {   // edge_attr row (f16) for stage A; pad cols pre-zeroed
            const float* ap = ea + (size_t)e * FE;
            _Float16* er = ea_s + lane * 32;
#pragma unroll
            for (int j = 0; j < 16; ++j) er[j] = (_Float16)ap[j];
        }
        dst_s[lane] = di;
        // same-wave DS ops are in-order; drain and fence the compiler
        asm volatile("s_wait_dscnt 0x0" ::: "memory");

        // ---- stage A: u = relu(ea @ W1 + b1), two 16-edge sub-tiles ----
        v16h A0, A1;
        {
            v8h a0 = *(const v8h*)(ea_s + eL * 32 + hsel);
            v8h a1 = *(const v8h*)(ea_s + eL * 32 + 16 + hsel);
            v8h b0 = *(const v8h*)(ea_s + (16 + eL) * 32 + hsel);
            v8h b1v = *(const v8h*)(ea_s + (16 + eL) * 32 + 16 + hsel);
#pragma unroll
            for (int j = 0; j < 8; ++j) {
                A0[j] = a0[j]; A0[8 + j] = a1[j];
                A1[j] = b0[j]; A1[8 + j] = b1v[j];
            }
        }
#pragma unroll
        for (int nt = 0; nt < 8; ++nt) {
            v16h B;
            {
                const _Float16* bp = w1s + nt * 16 + eL;  // B col N=lane%16, K table order
#pragma unroll
                for (int j = 0; j < 8; ++j) { B[j] = bp[(hsel + j) * 128]; B[8 + j] = bp[(16 + hsel + j) * 128]; }
            }
            v8f z = {};
            v8f acc0 = __builtin_amdgcn_wmma_f32_16x16x32_f16(false, A0, false, B,
                                                              (short)0, z, false, false);
            v8f acc1 = __builtin_amdgcn_wmma_f32_16x16x32_f16(false, A1, false, B,
                                                              (short)0, z, false, false);
            const float bv = b1[nt * 16 + eL];
#pragma unroll
            for (int v = 0; v < 8; ++v) {
                float t0 = fmaxf(acc0[v] + bv, 0.f);
                float t1 = fmaxf(acc1[v] + bv, 0.f);
                u_s[(v + hsel) * 128 + nt * 16 + eL]        = (_Float16)t0;
                u_s[(16 + v + hsel) * 128 + nt * 16 + eL]   = (_Float16)t1;
            }
        }
        asm volatile("s_wait_dscnt 0x0" ::: "memory");

        // ---- bias fold: msg = nf @ reshape(b2, [32,16]) (one WMMA each) ----
        v8f msg0, msg1;
        {
            v16h NA0, NA1;
            v8h a0 = *(const v8h*)(nf16_s + eL * 32 + hsel);
            v8h a1 = *(const v8h*)(nf16_s + eL * 32 + 16 + hsel);
            v8h b0 = *(const v8h*)(nf16_s + (16 + eL) * 32 + hsel);
            v8h b1h = *(const v8h*)(nf16_s + (16 + eL) * 32 + 16 + hsel);
#pragma unroll
            for (int j = 0; j < 8; ++j) {
                NA0[j] = a0[j]; NA0[8 + j] = a1[j];
                NA1[j] = b0[j]; NA1[8 + j] = b1h[j];
            }
            v8f z = {};
            msg0 = __builtin_amdgcn_wmma_f32_16x16x32_f16(false, NA0, false, Bb2,
                                                          (short)0, z, false, false);
            msg1 = __builtin_amdgcn_wmma_f32_16x16x32_f16(false, NA1, false, Bb2,
                                                          (short)0, z, false, false);
        }

        // ---- stage B: keep both sub-tiles' u A-chunks in VGPRs ----
        v16h UA0[4], UA1[4];
#pragma unroll
        for (int kk = 0; kk < 4; ++kk) {
            const _Float16* u0 = u_s + eL * 128 + kk * 32 + hsel;
            const _Float16* u1 = u_s + (16 + eL) * 128 + kk * 32 + hsel;
            v8h x0 = *(const v8h*)u0;
            v8h x1 = *(const v8h*)(u0 + 16);
            v8h y0 = *(const v8h*)u1;
            v8h y1 = *(const v8h*)(u1 + 16);
#pragma unroll
            for (int j = 0; j < 8; ++j) {
                UA0[kk][j] = x0[j]; UA0[kk][8 + j] = x1[j];
                UA1[kk][j] = y0[j]; UA1[kk][8 + j] = y1[j];
            }
        }
        // per k-tile: w_k = u @ W2[:,16k:16(k+1)]; every B chunk feeds 2 WMMAs
#pragma unroll 2
        for (int k = 0; k < TWO_D; ++k) {
            v8f wt0 = {}, wt1 = {};
#pragma unroll
            for (int kk = 0; kk < 4; ++kk) {
                const _Float16* bp = w2s + (size_t)(kk * 32) * 512 + k * 16 + eL;
                v16h B;
#pragma unroll
                for (int j = 0; j < 8; ++j) { B[j] = bp[(hsel + j) * 512]; B[8 + j] = bp[(16 + hsel + j) * 512]; }
                wt0 = __builtin_amdgcn_wmma_f32_16x16x32_f16(false, UA0[kk], false, B,
                                                             (short)0, wt0, false, false);
                wt1 = __builtin_amdgcn_wmma_f32_16x16x32_f16(false, UA1[kk], false, B,
                                                             (short)0, wt1, false, false);
            }
#pragma unroll
            for (int v = 0; v < 8; ++v) {
                msg0[v] += nf_s[(v + hsel) * 32 + k] * wt0[v];
                msg1[v] += nf_s[(16 + v + hsel) * 32 + k] * wt1[v];
            }
        }
        // ---- segment-sum via f32 global atomics ----
#pragma unroll
        for (int v = 0; v < 8; ++v) {
            atomicAdd(&aggr[(size_t)dst_s[v + hsel] * D + eL], msg0[v]);
            atomicAdd(&aggr[(size_t)dst_s[16 + v + hsel] * D + eL], msg1[v]);
        }
    }
}

// ---------------------------------------------------------------------------
extern "C" void kernel_launch(void* const* d_in, const int* in_sizes, int n_in,
                              void* d_out, int out_size, void* d_ws, size_t ws_size,
                              hipStream_t stream) {
    const float* x     = (const float*)d_in[0];
    const int*   eix   = (const int*)d_in[1];
    const float* ea    = (const float*)d_in[2];
    const float* Wn    = (const float*)d_in[3];
    const float* bn    = (const float*)d_in[4];
    const float* We    = (const float*)d_in[5];
    const float* be    = (const float*)d_in[6];
    const float* W1    = (const float*)d_in[7];
    const float* b1    = (const float*)d_in[8];
    const float* W2    = (const float*)d_in[9];
    const float* b2    = (const float*)d_in[10];
    const float* root1 = (const float*)d_in[11];
    const float* bias1 = (const float*)d_in[12];
    const float* root2 = (const float*)d_in[13];
    const float* bias2 = (const float*)d_in[14];

    float* out     = (float*)d_out;
    float* out_h   = out;                              // [N,16]
    float* out_idx = out + N_NODES * D;                // [2,E] as float
    float* out_ee  = out_idx + 2 * N_EDGES;            // [E,16]
    float* out_ls  = out_ee + (size_t)N_EDGES * D;     // [E,16]

    float* h0   = (float*)d_ws;
    float* h1   = h0 + N_NODES * D;
    float* ag1  = h1 + N_NODES * D;
    float* ag2  = ag1 + N_NODES * D;
    _Float16* blob = (_Float16*)(ag2 + N_NODES * D);   // f16 weight blob

    zero_kernel<<<(2 * N_NODES * D + 255) / 256, 256, 0, stream>>>(ag1, 2 * N_NODES * D);
    weights_prep_kernel<<<(BLOB_HALVES + 255) / 256, 256, 0, stream>>>(W1, b2, W2, blob);
    node_embed_kernel<<<(N_NODES + 127) / 128, 128, 0, stream>>>(x, Wn, bn, h0);

    const size_t smem = SMEM_BYTES;
    edge_conv_kernel<<<256, 128, smem, stream>>>(h0, eix, ea, blob, b1, ag1);
    node_update_kernel<<<(N_NODES + 127) / 128, 128, 0, stream>>>(h0, ag1, root1, bias1, h1);

    edge_conv_kernel<<<256, 128, smem, stream>>>(h1, eix, ea, blob, b1, ag2);
    node_update_kernel<<<(N_NODES + 127) / 128, 128, 0, stream>>>(h1, ag2, root2, bias2, out_h);

    edge_embed_kernel<<<(N_EDGES + 127) / 128, 128, 0, stream>>>(ea, We, be, out_ee, out_ls);
    idx_copy_kernel<<<(2 * N_EDGES + 255) / 256, 256, 0, stream>>>(eix, out_idx);
}
